// FLAHGRN2_5351529251279
// MI455X (gfx1250) — matrix operand order repaired and checked by reading, not compiled, not run on hardware
//
#include <hip/hip_runtime.h>
#include <hip/hip_bf16.h>
#include <cstdint>

// ---------- types ----------
typedef __attribute__((ext_vector_type(16))) __bf16 bf16x16;
typedef __attribute__((ext_vector_type(8)))  __bf16 bf16x8;
typedef __attribute__((ext_vector_type(8)))  float  f32x8;

#if defined(__AMDGCN__)
typedef int v4i_vs __attribute__((vector_size(16)));                      // builtin's V4i
typedef __attribute__((address_space(1))) v4i_vs as1_v4i;                 // global
typedef __attribute__((address_space(3))) v4i_vs as3_v4i;                 // LDS
#endif

// ---------- async global->LDS copy (CDNA5 GLOBAL_LOAD_ASYNC_TO_LDS_B128) ----------
#if defined(__AMDGCN__) && __has_builtin(__builtin_amdgcn_global_load_async_to_lds_b128)
#define HAVE_ASYNC_COPY 1
#else
#define HAVE_ASYNC_COPY 0
#endif

__device__ __forceinline__ void async_copy16(const __bf16* gsrc, __bf16* ldst) {
#if HAVE_ASYNC_COPY
    __builtin_amdgcn_global_load_async_to_lds_b128(
        (as1_v4i*)(uintptr_t)gsrc,
        (as3_v4i*)(uint32_t)(uintptr_t)ldst,
        0, 0);
#else
    *(bf16x8*)ldst = *(const bf16x8*)gsrc;
#endif
}

__device__ __forceinline__ void wait_async0() {
#if defined(__AMDGCN__)
#if __has_builtin(__builtin_amdgcn_s_wait_asynccnt)
    __builtin_amdgcn_s_wait_asynccnt(0);
#elif HAVE_ASYNC_COPY
    asm volatile("s_wait_asynccnt 0x0" ::: "memory");
#endif
#endif
}

// ---------- WMMA helpers (CDNA5 16x16x32 bf16, f32 accum) ----------
__device__ __forceinline__ f32x8 wmma_bf16(bf16x16 a, bf16x16 b, f32x8 c) {
    // (neg_a, A, neg_b, B, c_mod, C, reuse_a, reuse_b)
    return __builtin_amdgcn_wmma_f32_16x16x32_bf16(false, a, false, b, (short)0, c, false, false);
}

// A-matrix 16x32 (MxK) per-lane layout: lanes 0-15 -> K{0..7,16..23},
// lanes 16-31 -> K{8..15,24..31}; row = lane&15 handled by caller.
__device__ __forceinline__ bf16x16 frag_a(const __bf16* row_kk, int lane) {
    const int koff = (lane & 16) ? 8 : 0;
    union { bf16x16 v; bf16x8 h[2]; } u;
    u.h[0] = *(const bf16x8*)(row_kk + koff);
    u.h[1] = *(const bf16x8*)(row_kk + 16 + koff);
    return u.v;
}
// B-matrix 32x16 (KxN): lane = N, lanes 0-15 hold K{0..7,16..23}, lanes 16-31 K{8..15,24..31}.
// Caller passes pointer to K-contiguous row for column n (B stored transposed/K-major).
__device__ __forceinline__ bf16x16 frag_b(const __bf16* col_kk, int lane) {
    const int koff = (lane & 16) ? 8 : 0;
    union { bf16x16 v; bf16x8 h[2]; } u;
    u.h[0] = *(const bf16x8*)(col_kk + koff);
    u.h[1] = *(const bf16x8*)(col_kk + 16 + koff);
    return u.v;
}

// ---------- f32 -> bf16 conversion ----------
__global__ __launch_bounds__(256) void cvt_bf16_kernel(const float* __restrict__ src,
                                                       __bf16* __restrict__ dst, size_t n) {
    size_t i = (size_t)blockIdx.x * blockDim.x + threadIdx.x;
    size_t stride = (size_t)gridDim.x * blockDim.x;
    for (; i < n; i += stride) dst[i] = (__bf16)src[i];
}

// ---------- WMMA GEMM: C[m,n] = epilogue( sum_k A[m,k] * W[n,k] ) ----------
// A: [M][K] bf16 row-major, W: [N][K] bf16 row-major (i.e. x @ W^T form).
// mode 0: identity, 1: sigmoid. Double-buffered LDS stages filled via async copies.
__global__ __launch_bounds__(256) void wmma_gemm_kernel(const __bf16* __restrict__ A,
                                                        const __bf16* __restrict__ W,
                                                        float* __restrict__ C,
                                                        int M, int N, int K, int mode) {
    __shared__ __align__(16) __bf16 sA[2][128][72];  // M x K stages
    __shared__ __align__(16) __bf16 sB[2][64][72];   // N x K stages (K-contiguous per n row)
    const int tid  = threadIdx.x;
    const int lane = tid & 31;
    const int wave = tid >> 5;
    const int wm = (wave & 3) * 32;   // wave M offset within block tile
    const int wn = (wave >> 2) * 32;  // wave N offset
    const int m0 = blockIdx.y * 128;
    const int n0 = blockIdx.x * 64;

    f32x8 acc[2][2] = {};

    // stage issuer: 128x64 A chunk (4 x b128/thread) + 64x64 W chunk (2 x b128/thread)
    auto issue_stage = [&](int buf, int k0) {
#pragma unroll
        for (int c = 0; c < 4; ++c) {
            int linear = tid * 4 + c;          // 0..1023
            int row = linear >> 3;             // 8 chunks per row
            int kc  = (linear & 7) * 8;
            async_copy16(&A[(size_t)(m0 + row) * K + k0 + kc], &sA[buf][row][kc]);
        }
#pragma unroll
        for (int c = 0; c < 2; ++c) {
            int linear = tid * 2 + c;          // 0..511
            int row = linear >> 3;
            int kc  = (linear & 7) * 8;
            async_copy16(&W[(size_t)(n0 + row) * K + k0 + kc], &sB[buf][row][kc]);
        }
    };

    issue_stage(0, 0);

    for (int k0 = 0; k0 < K; k0 += 64) {
        const int buf = (k0 >> 6) & 1;
        wait_async0();            // own stage-`buf` copies have landed in LDS
        __syncthreads();          // all waves' copies visible
        if (k0 + 64 < K) issue_stage(buf ^ 1, k0 + 64);   // overlap next stage with compute
#pragma unroll
        for (int kk = 0; kk < 64; kk += 32) {
#pragma unroll
            for (int im = 0; im < 2; ++im) {
                bf16x16 af = frag_a(&sA[buf][wm + im * 16 + (lane & 15)][kk], lane);
#pragma unroll
                for (int in = 0; in < 2; ++in) {
                    bf16x16 bfg = frag_b(&sB[buf][wn + in * 16 + (lane & 15)][kk], lane);
                    acc[im][in] = wmma_bf16(af, bfg, acc[im][in]);
                }
            }
        }
        __syncthreads();          // finish reading `buf` before it is refilled
    }

    // epilogue: C/D layout -> VGPR r: M = r + (lane>=16 ? 8 : 0), N = lane&15
#pragma unroll
    for (int im = 0; im < 2; ++im) {
#pragma unroll
        for (int in = 0; in < 2; ++in) {
#pragma unroll
            for (int r = 0; r < 8; ++r) {
                int m = m0 + wm + im * 16 + r + ((lane & 16) ? 8 : 0);
                int n = n0 + wn + in * 16 + (lane & 15);
                float x = acc[im][in][r];
                if (mode == 1) x = 1.0f / (1.0f + __expf(-x));
                C[(size_t)m * N + n] = x;
            }
        }
    }
}

// ---------- GLA chunk kernel (one workgroup per (b,h), sequential over chunks) ----------
#define GLA_CH   64
#define GLA_DH   128
#define GC_OFF   0                               // float[64*128]
#define GL_OFF   (GC_OFF + 64 * 128 * 4)         // float[128]
#define EG_OFF   (GL_OFF + 512)                  // float[128]
#define QG_OFF   (EG_OFF + 512)                  // bf16[64*136]  (t x d)
#define KD_OFF   (QG_OFF + 64 * 136 * 2)         // bf16[64*136]  (s x d)
#define KGT_OFF  (KD_OFF + 64 * 136 * 2)         // bf16[128*72]  (d x t)
#define VCT_OFF  (KGT_OFF + 128 * 72 * 2)        // bf16[128*72]  (dv x t)
#define AS_OFF   (VCT_OFF + 128 * 72 * 2)        // bf16[64*72]   (t x s)
#define ST_OFF   (AS_OFF + 64 * 72 * 2)          // bf16[128*136] (dv x dk) = S^T
#define GLA_SMEM (ST_OFF + 128 * 136 * 2)        // 149504 bytes

__global__ __launch_bounds__(256) void gla_kernel(const float* __restrict__ Q,
                                                  const float* __restrict__ F,
                                                  const float* __restrict__ V,
                                                  float* __restrict__ O,
                                                  int Bsz, int L, int D, int H) {
    extern __shared__ char smem[];
    float*  gc   = (float*)(smem + GC_OFF);
    float*  gl   = (float*)(smem + GL_OFF);
    float*  egl  = (float*)(smem + EG_OFF);
    __bf16* qg   = (__bf16*)(smem + QG_OFF);
    __bf16* kd   = (__bf16*)(smem + KD_OFF);
    __bf16* kgT  = (__bf16*)(smem + KGT_OFF);
    __bf16* vcT  = (__bf16*)(smem + VCT_OFF);
    __bf16* As   = (__bf16*)(smem + AS_OFF);
    __bf16* ST   = (__bf16*)(smem + ST_OFF);

    const int tid  = threadIdx.x;
    const int lane = tid & 31;
    const int wave = tid >> 5;
    const int bh = blockIdx.x;
    const int b = bh / H, h = bh % H;
    const size_t baseoff = ((size_t)b * L) * D + (size_t)h * GLA_DH;
    const int nchunks = L / GLA_CH;

    // S state: wave w owns dk rows [w*16, w*16+16), 8 tiles across dv
    f32x8 Sacc[8] = {};
    for (int i = tid; i < 128 * 136; i += 256) ST[i] = (__bf16)0.0f;
    __syncthreads();

    for (int n = 0; n < nchunks; ++n) {
        const size_t rowbase = baseoff + (size_t)n * GLA_CH * D;

        // pass 1: per-dim inclusive cumsum of log(f) over the chunk
        if (tid < GLA_DH) {
            float c = 0.0f;
            for (int t = 0; t < GLA_CH; ++t) {
                float fv = F[rowbase + (size_t)t * D + tid];
                c += __logf(fv);
                gc[t * GLA_DH + tid] = c;
            }
            gl[tid]  = c;
            egl[tid] = __expf(c);
        }
        __syncthreads();

        // pass 2: build bf16 operand tiles
        for (int e = tid; e < GLA_CH * GLA_DH; e += 256) {
            int t = e >> 7, d = e & 127;
            size_t gidx = rowbase + (size_t)t * D + d;
            float fv = F[gidx];
            float qv = Q[gidx];
            float vv = V[gidx];
            float c  = gc[e];
            float gL = gl[d];
            float kv = 1.0f - fv;
            qg[t * 136 + d] = (__bf16)(qv * __expf(c));
            kd[t * 136 + d] = (__bf16)(kv * __expf(-c));
            kgT[d * 72 + t] = (__bf16)(kv * __expf(gL - c));
            vcT[d * 72 + t] = (__bf16)vv;
        }
        __syncthreads();

        // step B: A = qg @ kd^T (64x64), tril mask, store bf16
        {
            int tm = (wave & 3) * 16;
#pragma unroll
            for (int j = 0; j < 2; ++j) {
                int tn = (((wave >> 2) * 2) + j) * 16;
                f32x8 a = {};
#pragma unroll
                for (int kk = 0; kk < 128; kk += 32) {
                    bf16x16 af  = frag_a(&qg[(tm + (lane & 15)) * 136 + kk], lane);
                    bf16x16 bfg = frag_b(&kd[(tn + (lane & 15)) * 136 + kk], lane);
                    a = wmma_bf16(af, bfg, a);
                }
#pragma unroll
                for (int r = 0; r < 8; ++r) {
                    int t = tm + r + ((lane & 16) ? 8 : 0);
                    int s = tn + (lane & 15);
                    As[t * 72 + s] = (__bf16)((s <= t) ? a[r] : 0.0f);
                }
            }
        }
        __syncthreads();

        // step C: o = As @ vc + qg @ S_prev  (64 x 128)
        {
            int om = (wave & 3) * 16;
#pragma unroll
            for (int j = 0; j < 4; ++j) {
                int on = (((wave >> 2) * 4) + j) * 16;
                f32x8 o = {};
#pragma unroll
                for (int kk = 0; kk < 64; kk += 32) {      // intra, K = s
                    bf16x16 af  = frag_a(&As[(om + (lane & 15)) * 72 + kk], lane);
                    bf16x16 bfg = frag_b(&vcT[(on + (lane & 15)) * 72 + kk], lane);
                    o = wmma_bf16(af, bfg, o);
                }
#pragma unroll
                for (int kk = 0; kk < 128; kk += 32) {     // inter, K = dk
                    bf16x16 af  = frag_a(&qg[(om + (lane & 15)) * 136 + kk], lane);
                    bf16x16 bfg = frag_b(&ST[(on + (lane & 15)) * 136 + kk], lane);
                    o = wmma_bf16(af, bfg, o);
                }
#pragma unroll
                for (int r = 0; r < 8; ++r) {
                    int t  = om + r + ((lane & 16) ? 8 : 0);
                    int dv = on + (lane & 15);
                    O[rowbase + (size_t)t * D + dv] = o[r];
                }
            }
        }
        __syncthreads();

        // step D: S = S * exp(g_last) + kg^T @ vc
        {
            int sm = wave * 16;
#pragma unroll
            for (int j = 0; j < 8; ++j) {
#pragma unroll
                for (int r = 0; r < 8; ++r) {
                    int m = sm + r + ((lane & 16) ? 8 : 0);
                    Sacc[j][r] *= egl[m];
                }
                int sn = j * 16;
#pragma unroll
                for (int kk = 0; kk < 64; kk += 32) {      // K = t
                    bf16x16 af  = frag_a(&kgT[(sm + (lane & 15)) * 72 + kk], lane);
                    bf16x16 bfg = frag_b(&vcT[(sn + (lane & 15)) * 72 + kk], lane);
                    Sacc[j] = wmma_bf16(af, bfg, Sacc[j]);
                }
            }
        }
        // step E: dump S -> S^T bf16 for next chunk's B operand
        {
            int sm = wave * 16;
#pragma unroll
            for (int j = 0; j < 8; ++j) {
                int sn = j * 16;
#pragma unroll
                for (int r = 0; r < 8; ++r) {
                    int m  = sm + r + ((lane & 16) ? 8 : 0);
                    int nn = sn + (lane & 15);
                    ST[nn * 136 + m] = (__bf16)Sacc[j][r];
                }
            }
        }
        __syncthreads();
    }
}

// ---------- RMS norm (per row of D) -> bf16 ----------
__global__ __launch_bounds__(256) void rmsnorm_kernel(const float* __restrict__ X,
                                                      const float* __restrict__ w,
                                                      __bf16* __restrict__ Y, int D) {
    __shared__ float red[8];
    size_t row = blockIdx.x;
    const float* x = X + row * (size_t)D;
    float s = 0.0f;
    for (int i = threadIdx.x; i < D; i += 256) { float v = x[i]; s += v * v; }
#pragma unroll
    for (int off = 16; off > 0; off >>= 1) s += __shfl_down(s, off, 32);
    int lane = threadIdx.x & 31, wave = threadIdx.x >> 5;
    if (lane == 0) red[wave] = s;
    __syncthreads();
    if (wave == 0) {
        float t = (lane < 8) ? red[lane] : 0.0f;
#pragma unroll
        for (int off = 4; off > 0; off >>= 1) t += __shfl_down(t, off, 32);
        if (lane == 0) red[0] = t;
    }
    __syncthreads();
    float scale = rsqrtf(red[0] / (float)D + 1e-5f);
    for (int i = threadIdx.x; i < D; i += 256)
        Y[row * (size_t)D + i] = (__bf16)(x[i] * scale * w[i]);
}

// ---------- host launcher ----------
extern "C" void kernel_launch(void* const* d_in, const int* in_sizes, int n_in,
                              void* d_out, int out_size, void* d_ws, size_t ws_size,
                              hipStream_t stream) {
    (void)in_sizes; (void)n_in; (void)out_size; (void)ws_size;
    const float* x   = (const float*)d_in[0];
    const float* Wq  = (const float*)d_in[1];
    const float* Wf  = (const float*)d_in[2];
    const float* Wi  = (const float*)d_in[3];
    const float* Wo  = (const float*)d_in[4];
    const float* gnw = (const float*)d_in[5];

    const int B = 4, L = 2048, D = 2048, H = 16;
    const size_t Mrows = (size_t)B * L;      // 8192
    const size_t XE = Mrows * D;             // 16.7M elems
    const size_t WE = (size_t)D * D;

    char* ws = (char*)d_ws;
    __bf16* xb  = (__bf16*)ws;              ws += XE * 2;
    __bf16* wqb = (__bf16*)ws;              ws += WE * 2;
    __bf16* wfb = (__bf16*)ws;              ws += WE * 2;
    __bf16* wib = (__bf16*)ws;              ws += WE * 2;
    __bf16* wob = (__bf16*)ws;              ws += WE * 2;
    float*  qf  = (float*)ws;               ws += XE * 4;
    float*  ff  = (float*)ws;               ws += XE * 4;
    float*  vf  = (float*)ws;               ws += XE * 4;
    float*  of  = (float*)ws;               ws += XE * 4;
    __bf16* ob  = (__bf16*)ws;              ws += XE * 2;

    // 1) convert inputs to bf16
    cvt_bf16_kernel<<<4096, 256, 0, stream>>>(x,  xb,  XE);
    cvt_bf16_kernel<<<2048, 256, 0, stream>>>(Wq, wqb, WE);
    cvt_bf16_kernel<<<2048, 256, 0, stream>>>(Wf, wfb, WE);
    cvt_bf16_kernel<<<2048, 256, 0, stream>>>(Wi, wib, WE);
    cvt_bf16_kernel<<<2048, 256, 0, stream>>>(Wo, wob, WE);

    // 2) q = sigmoid(x Wq^T), f = sigmoid(x Wf^T), i = x Wi^T
    dim3 ggrid(D / 64, (int)(Mrows / 128));
    wmma_gemm_kernel<<<ggrid, 256, 0, stream>>>(xb, wqb, qf, (int)Mrows, D, D, 1);
    wmma_gemm_kernel<<<ggrid, 256, 0, stream>>>(xb, wfb, ff, (int)Mrows, D, D, 1);
    wmma_gemm_kernel<<<ggrid, 256, 0, stream>>>(xb, wib, vf, (int)Mrows, D, D, 0);

    // 3) chunked gated linear attention, one workgroup per (b,h)
    gla_kernel<<<B * H, 256, GLA_SMEM, stream>>>(qf, ff, vf, of, B, L, D, H);

    // 4) RMS norm -> bf16
    rmsnorm_kernel<<<(int)Mrows, 256, 0, stream>>>(of, gnw, ob, D);

    // 5) out = o_norm @ Wo^T (f32 result straight to d_out)
    wmma_gemm_kernel<<<ggrid, 256, 0, stream>>>(ob, wob, (float*)d_out, (int)Mrows, D, D, 0);
}
